// GenerativeNeSyMNISTAdd_82849919140089
// MI455X (gfx1250) — matrix operand order repaired and checked by reading, not compiled, not run on hardware
//
#include <hip/hip_runtime.h>
#include <hip/hip_fp16.h>

typedef _Float16 h16;
typedef __attribute__((ext_vector_type(16))) _Float16 v16h;
typedef __attribute__((ext_vector_type(8)))  float    v8f;

#define BATCH 2048
#define NDIG  4
#define NVAL  10
#define MTOT  10000     // 625 tiles of 16
#define MTILES 625
#define KPAD  64        // 4 digits * 16 padded value slots
#define WAVES 8
#define EPSF  1e-10f

// ---------------------------------------------------------------------------
// Kernel 1a: per batch row b: log(softmax(logits)+eps) -> f16 A rows (B x 64),
// plus 3-digit target code for the mask compare.
// ---------------------------------------------------------------------------
__global__ void prep_a(const float* __restrict__ logits,
                       const int*  __restrict__ yBY,
                       h16*  __restrict__ Alogp,
                       int*  __restrict__ codeB) {
  int b = blockIdx.x * blockDim.x + threadIdx.x;
  if (b >= BATCH) return;
  const float* L = logits + b * NDIG * NVAL;
  h16* out = Alogp + b * KPAD;
  #pragma unroll
  for (int n = 0; n < NDIG; ++n) {
    float m = -1e30f;
    #pragma unroll
    for (int d = 0; d < NVAL; ++d) m = fmaxf(m, L[n * NVAL + d]);
    float s = 0.f;
    #pragma unroll
    for (int d = 0; d < NVAL; ++d) s += __expf(L[n * NVAL + d] - m);
    float lse = m + __logf(s);
    #pragma unroll
    for (int d = 0; d < NVAL; ++d) {
      float p = __expf(L[n * NVAL + d] - lse);          // softmax prob
      out[n * 16 + d] = (h16)__logf(p + EPSF);          // log(p + eps)
    }
    #pragma unroll
    for (int d = NVAL; d < 16; ++d) out[n * 16 + d] = (h16)0.f; // padded slots
  }
  codeB[b] = yBY[b * 3] * 100 + yBY[b * 3 + 1] * 10 + yBY[b * 3 + 2];
}

// ---------------------------------------------------------------------------
// Kernel 1b: per assignment m: one-hot row (M x 64) in f16 (4 ones), plus
// 3-digit sum code.
// ---------------------------------------------------------------------------
__global__ void prep_b(const int* __restrict__ allW,
                       const int* __restrict__ allY,
                       h16* __restrict__ BmatT,
                       int* __restrict__ codeM) {
  int m = blockIdx.x * blockDim.x + threadIdx.x;
  if (m >= MTOT) return;
  h16 row[KPAD];
  #pragma unroll
  for (int k = 0; k < KPAD; ++k) row[k] = (h16)0.f;
  for (int n = 0; n < NDIG; ++n) row[n * 16 + allW[m * NDIG + n]] = (h16)1.f;
  uint4* dst = (uint4*)(BmatT + m * KPAD);
  const uint4* src = (const uint4*)row;
  #pragma unroll
  for (int i = 0; i < 8; ++i) dst[i] = src[i];
  codeM[m] = allY[m * 3] * 100 + allY[m * 3 + 1] * 10 + allY[m * 3 + 2];
}

// ---------------------------------------------------------------------------
// Kernel 2: fused WMMA GEMM + masked reductions.
// Block = 16 batch rows, 8 waves stride the 625 M-tiles.
// log_probs tile = A(16x64 f16) x OneHot^T(16x64 f16)^T via two
// v_wmma_f32_16x16x32_f16 (K = 0..31, 32..63).
// ---------------------------------------------------------------------------
__global__ void __launch_bounds__(256)
gemm_reduce(const h16* __restrict__ Alogp, const h16* __restrict__ BmatT,
            const int* __restrict__ codeB, const int* __restrict__ codeM,
            float* __restrict__ lossB, float* __restrict__ entB) {
  __shared__ float sh0[WAVES][16];
  __shared__ float sh1[WAVES][16];

  const int tid  = threadIdx.x;
  const int wave = tid >> 5;
  const int lane = tid & 31;
  const int half = lane >> 4;   // 0: lanes 0-15, 1: lanes 16-31
  const int l16  = lane & 15;
  const int bbase = blockIdx.x * 16;

  // --- A fragments (loop invariant), ISA 16-bit A 16x32 layout ---
  // lanes 0-15 row M=l16: e0..7 = K0..7,  e8..15 = K16..23
  // lanes16-31 row M=l16: e0..7 = K8..15, e8..15 = K24..31
  union frag { v16h v; uint4 u[2]; };
  frag a0, a1;
  const h16* arow = Alogp + (size_t)(bbase + l16) * KPAD + half * 8;
  a0.u[0] = *(const uint4*)(arow +  0);
  a0.u[1] = *(const uint4*)(arow + 16);
  a1.u[0] = *(const uint4*)(arow + 32);
  a1.u[1] = *(const uint4*)(arow + 48);

  // Target codes for the 8 batch rows this lane-half accumulates (D-layout).
  int cb[8];
  #pragma unroll
  for (int v = 0; v < 8; ++v) cb[v] = codeB[bbase + half * 8 + v];

  float s0[8] = {0.f, 0.f, 0.f, 0.f, 0.f, 0.f, 0.f, 0.f}; // sum p*mask
  float s1[8] = {0.f, 0.f, 0.f, 0.f, 0.f, 0.f, 0.f, 0.f}; // sum p*mask*log p

  for (int mt = wave; mt < MTILES; mt += WAVES) {
    const int mg = mt * 16 + l16;
    // --- B fragment, ISA 16-bit B 32x16 layout: lane = column (assignment),
    //     lanes 0-15 hold K=0..7,16..23, lanes 16-31 hold K=8..15,24..31 (matches A).
    const h16* brow = BmatT + (size_t)mg * KPAD + half * 8;
    frag b0, b1;
    b0.u[0] = *(const uint4*)(brow +  0);
    b0.u[1] = *(const uint4*)(brow + 16);
    b1.u[0] = *(const uint4*)(brow + 32);
    b1.u[1] = *(const uint4*)(brow + 48);

    // prefetch next M-tile's one-hot row (global_prefetch_b8)
    int mtn = mt + WAVES;
    if (mtn < MTILES)
      __builtin_prefetch(BmatT + (size_t)(mtn * 16 + l16) * KPAD, 0, 0);

    v8f c = {};
    c = __builtin_amdgcn_wmma_f32_16x16x32_f16(false, a0.v, false, b0.v,
                                               (short)0, c, false, false);
    c = __builtin_amdgcn_wmma_f32_16x16x32_f16(false, a1.v, false, b1.v,
                                               (short)0, c, false, false);

    const int cm = codeM[mg];
    #pragma unroll
    for (int v = 0; v < 8; ++v) {
      float lp = c[v];                 // log_probs[b = bbase+half*8+v, m = mg]
      float p  = __expf(lp);
      if (cm == cb[v]) { s0[v] += p; s1[v] += p * lp; }
    }
  }

  // Reduce across the 16 lanes of each half (deterministic shuffle tree).
  #pragma unroll
  for (int v = 0; v < 8; ++v) {
    #pragma unroll
    for (int off = 1; off < 16; off <<= 1) {
      s0[v] += __shfl_xor(s0[v], off, 16);
      s1[v] += __shfl_xor(s1[v], off, 16);
    }
  }
  if (l16 == 0) {
    #pragma unroll
    for (int v = 0; v < 8; ++v) {
      sh0[wave][half * 8 + v] = s0[v];
      sh1[wave][half * 8 + v] = s1[v];
    }
  }
  __syncthreads();

  if (tid < 16) {
    float S0 = 0.f, S1 = 0.f;
    #pragma unroll
    for (int w = 0; w < WAVES; ++w) { S0 += sh0[w][tid]; S1 += sh1[w][tid]; }
    float py  = S0 + EPSF;
    float lpy = __logf(py);
    lossB[bbase + tid] = -lpy;
    // H_cond = -sum q log q / 4 = -(S1 - S0*log(py)) / py / 4
    entB[bbase + tid] = -(S1 - S0 * lpy) / py * 0.25f;
  }
}

// ---------------------------------------------------------------------------
// Kernel 3: scalar loss = mean(lossB) - 0.1*mean(entB)
// ---------------------------------------------------------------------------
__global__ void finalize(const float* __restrict__ lossB,
                         const float* __restrict__ entB,
                         float* __restrict__ out) {
  __shared__ float sh[256];
  int t = threadIdx.x;
  float acc = 0.f;
  for (int b = t; b < BATCH; b += 256) acc += lossB[b] - 0.1f * entB[b];
  sh[t] = acc;
  __syncthreads();
  for (int s = 128; s > 0; s >>= 1) {
    if (t < s) sh[t] += sh[t + s];
    __syncthreads();
  }
  if (t == 0) out[0] = sh[0] / (float)BATCH;
}

extern "C" void kernel_launch(void* const* d_in, const int* in_sizes, int n_in,
                              void* d_out, int out_size, void* d_ws, size_t ws_size,
                              hipStream_t stream) {
  const float* logits = (const float*)d_in[0];  // (B,4,10) f32
  const int*   yBY    = (const int*)d_in[1];    // (B,3)    i32
  const int*   allW   = (const int*)d_in[2];    // (M,4)    i32
  const int*   allY   = (const int*)d_in[3];    // (M,3)    i32

  char* ws = (char*)d_ws;
  size_t off = 0;
  auto alloc = [&](size_t bytes) -> char* {
    char* p = ws + off;
    off = (off + bytes + 255) & ~(size_t)255;
    return p;
  };
  h16*   Alogp = (h16*)alloc((size_t)BATCH * KPAD * sizeof(h16));
  h16*   BmatT = (h16*)alloc((size_t)MTOT  * KPAD * sizeof(h16));
  int*   codeB = (int*)alloc((size_t)BATCH * sizeof(int));
  int*   codeM = (int*)alloc((size_t)MTOT  * sizeof(int));
  float* lossB = (float*)alloc((size_t)BATCH * sizeof(float));
  float* entB  = (float*)alloc((size_t)BATCH * sizeof(float));
  (void)ws_size; (void)in_sizes; (void)n_in; (void)out_size;

  hipLaunchKernelGGL(prep_a, dim3((BATCH + 255) / 256), dim3(256), 0, stream,
                     logits, yBY, Alogp, codeB);
  hipLaunchKernelGGL(prep_b, dim3((MTOT + 255) / 256), dim3(256), 0, stream,
                     allW, allY, BmatT, codeM);
  hipLaunchKernelGGL(gemm_reduce, dim3(BATCH / 16), dim3(256), 0, stream,
                     Alogp, BmatT, codeB, codeM, lossB, entB);
  hipLaunchKernelGGL(finalize, dim3(1), dim3(256), 0, stream,
                     lossB, entB, (float*)d_out);
}
